// Attention_50354196578441
// MI455X (gfx1250) — compile-verified, matrix operand order checked by reading, not run on hardware
//
#include <hip/hip_runtime.h>
#include <hip/hip_bf16.h>

// ---- problem constants (B=1) ----
#define S_LEN   2048
#define NHEADS  16
#define NKV     4
#define HEADDIM 128
#define HID_    2048
#define WIN_SZ  1024

typedef __bf16 bf16;
typedef __attribute__((ext_vector_type(16))) __bf16 v16bf;
typedef __attribute__((ext_vector_type(8)))  __bf16 v8bf;
typedef __attribute__((ext_vector_type(8)))  float  v8f;
typedef __attribute__((ext_vector_type(4)))  int    v4i;

// ---- CDNA5 async global->LDS path (guarded so both toolchains compile) ----
#if defined(__HIP_DEVICE_COMPILE__) && __has_builtin(__builtin_amdgcn_global_load_async_to_lds_b128)
#define HAVE_ASYNC_LDS 1
#else
#define HAVE_ASYNC_LDS 0
#endif

// Builtin signature (from hipcc diagnostic): param0 = v4i in AS1 (global,
// printed as "__device__"), param1 = v4i in AS3 (LDS), then imm offset, imm cpol.
typedef __attribute__((address_space(1))) v4i gv4i;
typedef __attribute__((address_space(3))) v4i lv4i;

__device__ __forceinline__ gv4i* gptr4(const void* p) {
  return (gv4i*)(unsigned long long)p;
}
__device__ __forceinline__ lv4i* lptr4(void* p) {
  // low 32 bits of a generic LDS address are the LDS offset (aperture scheme)
  return (lv4i*)(unsigned int)(unsigned long long)p;
}
__device__ __forceinline__ void wait_async0() {
#if __has_builtin(__builtin_amdgcn_s_wait_asynccnt)
  __builtin_amdgcn_s_wait_asynccnt(0);
#else
  asm volatile("s_wait_asynccnt 0x0" ::: "memory");
#endif
}

// Load one 16-bit WMMA operand fragment for this lane.
// Caller passes p = row_base + half*8; elements are K = {half*8+0..7, 16+half*8+0..7}.
__device__ __forceinline__ v16bf ld_frag(const bf16* p) {
  v8bf lo = *(const v8bf*)(p);
  v8bf hi = *(const v8bf*)(p + 16);
  v16bf r;
#pragma unroll
  for (int i = 0; i < 8; ++i) { r[i] = lo[i]; r[i + 8] = hi[i]; }
  return r;
}

__device__ __forceinline__ v8f wmma_bf16(v16bf a, v16bf b, v8f c) {
  return __builtin_amdgcn_wmma_f32_16x16x32_bf16(false, a, false, b, (short)0, c,
                                                 false, false);
}

// ---------------------------------------------------------------------------
// C[M,N] = A[M,K] (row-major, fp32 or bf16) @ W[N,K]^T (fp32 weights).
// 128x128x32 tiles, 256 threads = 8 waves in a 2x4 grid; each wave computes a
// 64x32 region = 4x2 WMMA tiles with f32 accumulation. bf16 A-tiles are staged
// with the CDNA5 async global->LDS engine (no VGPR round-trip).
// ---------------------------------------------------------------------------
template <typename AT>
__global__ __launch_bounds__(256) void gemm_wmma(const AT* __restrict__ A,
                                                 const float* __restrict__ W,
                                                 float* __restrict__ C,
                                                 int M, int N, int K) {
  constexpr int LDT = 40;                    // 32 + 8 bf16 pad: 80B rows, 16B aligned
  __shared__ bf16 As[128 * LDT];
  __shared__ bf16 Bs[128 * LDT];
  const int bm = blockIdx.y * 128;
  const int bn = blockIdx.x * 128;
  const int t    = threadIdx.x;
  const int lane = t & 31;
  const int wave = t >> 5;
  const int wr   = wave >> 2;                // 0..1
  const int wc   = wave & 3;                 // 0..3
  const int idx  = lane & 15;
  const int half = lane >> 4;
  const int lrow = t >> 1;                   // 0..127
  const int lcol = (t & 1) * 16;             // 0 or 16

  const v8f vz = {0.f, 0.f, 0.f, 0.f, 0.f, 0.f, 0.f, 0.f};
  v8f acc[4][2];
#pragma unroll
  for (int mi = 0; mi < 4; ++mi)
#pragma unroll
    for (int ni = 0; ni < 2; ++ni) acc[mi][ni] = vz;

  for (int k0 = 0; k0 < K; k0 += 32) {
    // ---- stage A tile ----
    {
      bf16* la = As + lrow * LDT + lcol;
      if constexpr (sizeof(AT) == 4) {
        // fp32 source: convert to bf16 through registers
        const float* ga = (const float*)A + (size_t)(bm + lrow) * K + k0 + lcol;
        v8bf a0, a1;
#pragma unroll
        for (int i = 0; i < 8; ++i) a0[i] = (bf16)ga[i];
#pragma unroll
        for (int i = 0; i < 8; ++i) a1[i] = (bf16)ga[8 + i];
        *(v8bf*)(la)     = a0;
        *(v8bf*)(la + 8) = a1;
        if (k0 + 32 < K) __builtin_prefetch(ga + 32, 0, 1);
      } else {
        // bf16 source: pure byte copy -> async global->LDS (ASYNCcnt path)
        const bf16* ga = (const bf16*)A + (size_t)(bm + lrow) * K + k0 + lcol;
#if HAVE_ASYNC_LDS
        __builtin_amdgcn_global_load_async_to_lds_b128(gptr4(ga), lptr4(la), 0, 0);
        __builtin_amdgcn_global_load_async_to_lds_b128(gptr4(ga + 8), lptr4(la + 8),
                                                       0, 0);
#else
        *(v8bf*)(la)     = *(const v8bf*)(ga);
        *(v8bf*)(la + 8) = *(const v8bf*)(ga + 8);
#endif
        if (k0 + 32 < K) __builtin_prefetch(ga + 32, 0, 1);
      }
      // ---- stage B tile: rows of W[N][K] (B = W^T columns), fp32 -> bf16 ----
      const float* gw = W + (size_t)(bn + lrow) * K + k0 + lcol;
      bf16* lb = Bs + lrow * LDT + lcol;
      v8bf w0, w1;
#pragma unroll
      for (int i = 0; i < 8; ++i) w0[i] = (bf16)gw[i];
#pragma unroll
      for (int i = 0; i < 8; ++i) w1[i] = (bf16)gw[8 + i];
      *(v8bf*)(lb)     = w0;
      *(v8bf*)(lb + 8) = w1;
      if (k0 + 32 < K) __builtin_prefetch(gw + 32, 0, 1);
    }
    if constexpr (sizeof(AT) == 2) {
#if HAVE_ASYNC_LDS
      wait_async0();                         // LDS writes visible before barrier
#endif
    }
    __syncthreads();

    v16bf af[4], bfr[2];
#pragma unroll
    for (int mi = 0; mi < 4; ++mi)
      af[mi] = ld_frag(As + (wr * 64 + mi * 16 + idx) * LDT + half * 8);
#pragma unroll
    for (int ni = 0; ni < 2; ++ni)
      bfr[ni] = ld_frag(Bs + (wc * 32 + ni * 16 + idx) * LDT + half * 8);
#pragma unroll
    for (int mi = 0; mi < 4; ++mi)
#pragma unroll
      for (int ni = 0; ni < 2; ++ni)
        acc[mi][ni] = wmma_bf16(af[mi], bfr[ni], acc[mi][ni]);
    __syncthreads();
  }

  // C/D layout: lane -> n = idx, VGPR v -> m = v + 8*half
#pragma unroll
  for (int mi = 0; mi < 4; ++mi)
#pragma unroll
    for (int ni = 0; ni < 2; ++ni) {
      const int n = bn + wc * 32 + ni * 16 + idx;
#pragma unroll
      for (int v = 0; v < 8; ++v) {
        const int m = bm + wr * 64 + mi * 16 + half * 8 + v;
        C[(size_t)m * N + n] = acc[mi][ni][v];
      }
    }
}

// ---------------------------------------------------------------------------
// Per-(s, head) RMSNorm + RoPE, fp32 -> bf16. One wave per head vector.
// Q gets the 1/sqrt(D) score scale folded in. Lane holds d = lane + {0,32,64,96}
// so both rotate-half partners (d, d+64) live in the same lane.
// ---------------------------------------------------------------------------
__global__ __launch_bounds__(256) void preproc_qk(
    const float* __restrict__ qf, const float* __restrict__ kf,
    const float* __restrict__ cosT, const float* __restrict__ sinT,
    const float* __restrict__ qw, const float* __restrict__ kw,
    bf16* __restrict__ qb, bf16* __restrict__ kb) {
  const int wid  = blockIdx.x * 8 + (threadIdx.x >> 5);
  const int lane = threadIdx.x & 31;
  const int NQ = S_LEN * NHEADS;
  int s, hh;
  const float* src; const float* w; bf16* dst; float oscale;
  if (wid < NQ) {
    s = wid / NHEADS; hh = wid - s * NHEADS;
    src = qf + (size_t)s * HID_ + hh * HEADDIM;
    dst = qb + ((size_t)hh * S_LEN + s) * HEADDIM;
    w = qw; oscale = 0.08838834764831845f;   // 1/sqrt(128)
  } else {
    const int j = wid - NQ;
    s = j / NKV; hh = j - s * NKV;
    src = kf + (size_t)s * (NKV * HEADDIM) + hh * HEADDIM;
    dst = kb + ((size_t)hh * S_LEN + s) * HEADDIM;
    w = kw; oscale = 1.0f;
  }
  float x[4];
#pragma unroll
  for (int i = 0; i < 4; ++i) x[i] = src[lane + 32 * i];
  float ss = x[0] * x[0] + x[1] * x[1] + x[2] * x[2] + x[3] * x[3];
#pragma unroll
  for (int m = 16; m >= 1; m >>= 1) ss += __shfl_xor(ss, m, 32);
  const float r = rsqrtf(ss * (1.0f / 128.0f) + 1e-5f);
  float xn[4];
#pragma unroll
  for (int i = 0; i < 4; ++i) xn[i] = x[i] * r * w[lane + 32 * i];
  const float rot[4] = {-xn[2], -xn[3], xn[0], xn[1]};
#pragma unroll
  for (int i = 0; i < 4; ++i) {
    const int d = lane + 32 * i;
    const float c  = cosT[(size_t)s * HEADDIM + d];
    const float sn = sinT[(size_t)s * HEADDIM + d];
    dst[d] = (bf16)((xn[i] * c + rot[i] * sn) * oscale);
  }
}

// V fp32 [s][kvh*D] -> bf16 transposed [kvh][d][s] (key axis contiguous, so
// P@V B-fragments are two 16B loads per lane).
__global__ __launch_bounds__(256) void v_transpose(const float* __restrict__ vf,
                                                   bf16* __restrict__ vt) {
  const int idx = blockIdx.x * 256 + threadIdx.x;
  const int g   = idx >> 18;                 // D*S = 262144 = 2^18
  const int rem = idx & ((1 << 18) - 1);
  const int d   = rem >> 11;                 // S = 2048 = 2^11
  const int s   = rem & (S_LEN - 1);
  vt[idx] = (bf16)vf[(size_t)s * (NKV * HEADDIM) + g * HEADDIM + d];
}

// ---------------------------------------------------------------------------
// Flash-style sliding-window attention with sink, one wave per (head, 16-q tile).
// 32 keys per block: scores = 8 WMMAs, P@V = 8 WMMAs. Online softmax per C-row
// (row stats are uniform across each 16-lane half after the butterfly).
// ---------------------------------------------------------------------------
__global__ __launch_bounds__(128) void attn_fused(
    const bf16* __restrict__ qb, const bf16* __restrict__ kb,
    const bf16* __restrict__ vt, const float* __restrict__ sinks,
    bf16* __restrict__ attn) {
  __shared__ bf16 p_sh[4][16 * 32];          // per-wave P transpose buffer
  const int wave = threadIdx.x >> 5;
  const int lane = threadIdx.x & 31;
  const int wid  = blockIdx.x * 4 + wave;
  const int h     = wid >> 7;                // S/16 = 128 q-tiles per head
  const int qbase = (wid & 127) * 16;
  const int g     = h >> 2;                  // H/KVH = 4
  const int idx   = lane & 15;
  const int half  = lane >> 4;
  bf16* pt = p_sh[wave];

  // Q fragments for this tile (scores are pre-scaled via preproc)
  v16bf qfr[4];
  {
    const bf16* qp = qb + ((size_t)h * S_LEN + qbase + idx) * HEADDIM + half * 8;
#pragma unroll
    for (int kc = 0; kc < 4; ++kc) qfr[kc] = ld_frag(qp + kc * 32);
  }

  const v8f vz = {0.f, 0.f, 0.f, 0.f, 0.f, 0.f, 0.f, 0.f};
  v8f o[8];
#pragma unroll
  for (int nt = 0; nt < 8; ++nt) o[nt] = vz;
  const float snk = sinks[h];
  float m_run[8], l_run[8];
#pragma unroll
  for (int v = 0; v < 8; ++v) { m_run[v] = snk; l_run[v] = 0.f; }

  int kstart = qbase - WIN_SZ;
  if (kstart < 0) kstart = 0;
  kstart &= ~31;
  const int kend = qbase + 15;

  for (int kbk = kstart; kbk <= kend; kbk += 32) {
    // ---- scores: Q(16x128) @ K^T -> two 16x16 tiles ----
    v8f sa[2] = {vz, vz};
#pragma unroll
    for (int kt = 0; kt < 2; ++kt) {
      int krow = kbk + kt * 16 + idx;
      if (krow > S_LEN - 1) krow = S_LEN - 1;    // masked below anyway
      const bf16* kp = kb + ((size_t)g * S_LEN + krow) * HEADDIM + half * 8;
#pragma unroll
      for (int kc = 0; kc < 4; ++kc)
        sa[kt] = wmma_bf16(qfr[kc], ld_frag(kp + kc * 32), sa[kt]);
    }
    // ---- mask + online softmax per row ----
    const int k0 = kbk + idx;
    const int k1 = kbk + 16 + idx;
#pragma unroll
    for (int v = 0; v < 8; ++v) {
      const int qrow = qbase + half * 8 + v;
      float s0 = sa[0][v];
      float s1 = sa[1][v];
      if (k0 > qrow || k0 < qrow - WIN_SZ) s0 = -3.0e38f;
      if (k1 > qrow || k1 < qrow - WIN_SZ) s1 = -3.0e38f;
      float mx = fmaxf(s0, s1);
#pragma unroll
      for (int m = 8; m >= 1; m >>= 1) mx = fmaxf(mx, __shfl_xor(mx, m, 16));
      const float mnew  = fmaxf(m_run[v], mx);
      const float alpha = __expf(m_run[v] - mnew);
      const float p0 = __expf(s0 - mnew);
      const float p1 = __expf(s1 - mnew);
      float rs = p0 + p1;
#pragma unroll
      for (int m = 8; m >= 1; m >>= 1) rs += __shfl_xor(rs, m, 16);
      l_run[v] = l_run[v] * alpha + rs;
      m_run[v] = mnew;
#pragma unroll
      for (int nt = 0; nt < 8; ++nt) o[nt][v] *= alpha;
      const int mrow = half * 8 + v;
      pt[mrow * 32 + idx]      = (bf16)p0;   // C-layout -> row-major P in LDS
      pt[mrow * 32 + 16 + idx] = (bf16)p1;
    }
    // ---- P(16x32) @ V(32x128): reload P as A-fragment, V from [g][d][s] ----
    const v16bf pf = ld_frag(pt + idx * 32 + half * 8);
#pragma unroll
    for (int nt = 0; nt < 8; ++nt) {
      const bf16* vp =
          vt + ((size_t)g * HEADDIM + nt * 16 + idx) * S_LEN + kbk + half * 8;
      o[nt] = wmma_bf16(pf, ld_frag(vp), o[nt]);
    }
  }

  // sink joins the denominator: softmax([scores, sink]) then drop sink column
  float inv[8];
#pragma unroll
  for (int v = 0; v < 8; ++v)
    inv[v] = 1.0f / (l_run[v] + __expf(snk - m_run[v]));
#pragma unroll
  for (int nt = 0; nt < 8; ++nt)
#pragma unroll
    for (int v = 0; v < 8; ++v) {
      const int mrow = qbase + half * 8 + v;
      attn[(size_t)mrow * HID_ + h * HEADDIM + nt * 16 + idx] =
          (bf16)(o[nt][v] * inv[v]);
    }
}

// ---------------------------------------------------------------------------
extern "C" void kernel_launch(void* const* d_in, const int* in_sizes, int n_in,
                              void* d_out, int out_size, void* d_ws, size_t ws_size,
                              hipStream_t stream) {
  const float* x     = (const float*)d_in[0];
  const float* cosT  = (const float*)d_in[1];
  const float* sinT  = (const float*)d_in[2];
  const float* wq    = (const float*)d_in[3];
  const float* wk    = (const float*)d_in[4];
  const float* wv    = (const float*)d_in[5];
  const float* wo    = (const float*)d_in[6];
  const float* sinks = (const float*)d_in[7];
  const float* qnw   = (const float*)d_in[8];
  const float* knw   = (const float*)d_in[9];
  float* out = (float*)d_out;

  char* ws = (char*)d_ws;                       // ~44 MB used
  float* qf = (float*)(ws);                     // [S][2048] fp32   16 MB
  float* kf = (float*)(ws + (16u << 20));       // [S][512]  fp32    4 MB
  float* vf = (float*)(ws + (20u << 20));       // [S][512]  fp32    4 MB
  bf16*  qb = (bf16*)(ws + (24u << 20));        // [H][S][D] bf16    8 MB
  bf16*  kb = (bf16*)(ws + (32u << 20));        // [KVH][S][D] bf16  2 MB
  bf16*  vt = (bf16*)(ws + (34u << 20));        // [KVH][D][S] bf16  2 MB (+4KB pad)
  bf16*  at = (bf16*)(ws + (36u << 20) + 4096); // [S][HID]  bf16    8 MB

  // QKV projections (x @ W^T), WMMA bf16 with f32 accumulate
  gemm_wmma<float><<<dim3(16, 16), 256, 0, stream>>>(x, wq, qf, 2048, 2048, 2048);
  gemm_wmma<float><<<dim3(4, 16),  256, 0, stream>>>(x, wk, kf, 2048, 512, 2048);
  gemm_wmma<float><<<dim3(4, 16),  256, 0, stream>>>(x, wv, vf, 2048, 512, 2048);

  // RMSNorm + RoPE + bf16 pack; V transpose to [kvh][d][s]
  preproc_qk<<<5120, 256, 0, stream>>>(qf, kf, cosT, sinT, qnw, knw, qb, kb);
  v_transpose<<<4096, 256, 0, stream>>>(vf, vt);

  // Sliding-window attention with sink (one wave per head x 16-query tile)
  attn_fused<<<512, 128, 0, stream>>>(qb, kb, vt, sinks, at);

  // Output projection
  gemm_wmma<bf16><<<dim3(16, 16), 256, 0, stream>>>(at, wo, out, 2048, 2048, 2048);
}
